// CNF_ODE_46093589021148
// MI455X (gfx1250) — compile-verified, matrix operand order checked by reading, not run on hardware
//
#include <hip/hip_runtime.h>

// Problem constants (from reference): N=524288 samples, D=2, H=128
#define D_IN   2
#define H      128
#define H2     64
#define WAVES  2
#define TPB    (WAVES * 32)    // 64 threads
#define SPW    32              // samples per wave: two 16-row WMMA tiles
#define SPB    (WAVES * SPW)   // samples per block = 64

typedef __attribute__((ext_vector_type(16))) _Float16 v16h;
typedef __attribute__((ext_vector_type(8)))  _Float16 v8h;
typedef __attribute__((ext_vector_type(8)))  float    v8f;

union V16U { struct { v8h lo, hi; } p; v16h v; };

// Load one 16x32 f16 fragment (A-layout; B uses the same lane pattern on a
// [col][k] stored matrix). Per-lane: halves 0..7 = row[k0 + 8*hi .. +7],
// halves 8..15 = row[k0 + 16 + 8*hi .. +7]  (two aligned ds_load_b128).
__device__ __forceinline__ v16h load_frag(const _Float16* base, int strideH,
                                          int row, int hi, int k0) {
  const _Float16* p = base + row * strideH + k0 + (hi << 3);
  V16U u;
  u.p.lo = *(const v8h*)p;
  u.p.hi = *(const v8h*)(p + 16);
  return u.v;
}

__device__ __forceinline__ v8f wmma16(v16h a, v16h b, v8f c) {
  return __builtin_amdgcn_wmma_f32_16x16x32_f16(false, a, false, b,
                                                (short)0, c, false, false);
}

// silu(z) and silu'(z) with raw v_rcp_f32 (no IEEE divide expansion)
__device__ __forceinline__ void silu_pair(float z, float& h, float& sp) {
  float e = __expf(-z);
  float s = __builtin_amdgcn_rcpf(1.0f + e);
  h  = z * s;
  sp = fmaf(h, 1.0f - s, s);   // s * (1 + z*(1-s))
}

__global__ __launch_bounds__(TPB) void cnf_ode_wmma_kernel(
    const float* __restrict__ yg, const float* __restrict__ tg,
    const float* __restrict__ vg,
    const float* __restrict__ W1g, const float* __restrict__ b1g,
    const float* __restrict__ W2g, const float* __restrict__ b2g,
    const float* __restrict__ W3g, const float* __restrict__ b3g,
    const float* __restrict__ W4g, const float* __restrict__ b4g,
    float* __restrict__ dyg, float* __restrict__ ndivg)
{
  // ---- weights (block-shared, both orientations for fwd/bwd B operands) ----
  __shared__ __align__(16) _Float16 sW2T[H][H];    // fwd L2:  [n][k] = W2[k][n]
  __shared__ __align__(16) _Float16 sW2 [H][H];    // bwd L2:  [k][n] = W2[k][n]
  __shared__ __align__(16) _Float16 sW3T[H2][H];   // fwd L3
  __shared__ __align__(16) _Float16 sW3 [H][H2];   // bwd L3
  __shared__ float sW1[3][H];
  __shared__ float sB1[H], sB2[H], sB3[H2];
  __shared__ float sW4[H2][2];
  __shared__ float sB4[2];
  // ---- per-wave staging (32 samples = 2 m-tiles per wave) ----
  __shared__ __align__(16) _Float16 habuf[WAVES][SPW][H];     // activations (A row-major)
  __shared__ __align__(16) _Float16 sp1b[WAVES][2][8][32][8]; // silu' raw C-tile order
  __shared__ __align__(16) _Float16 sp2b[WAVES][2][8][32][8];
  __shared__ __align__(16) _Float16 sp3b[WAVES][2][4][32][8];
  __shared__ float xbuf[WAVES][SPW][2];                       // y, then v
  __shared__ float accb[WAVES][SPW][2];                       // grad_y accumulators
  __shared__ float dyacc[WAVES][SPW][2];                      // dy accumulators (layer4)

  const int tid  = threadIdx.x;
  const int wave = tid >> 5;
  const int lane = tid & 31;
  const int lm   = lane & 15;   // C-layout: column within tile / A row within tile
  const int hi   = lane >> 4;   // C-layout: row half select

  // ================= weight preload (all threads) =================
  for (int i = tid; i < H * H; i += TPB) {
    int k = i >> 7, n = i & 127;
    _Float16 w = (_Float16)W2g[i];
    sW2[k][n] = w;  sW2T[n][k] = w;
  }
  for (int i = tid; i < H * H2; i += TPB) {
    int k = i / H2, n = i % H2;
    _Float16 w = (_Float16)W3g[i];
    sW3[k][n] = w;  sW3T[n][k] = w;
  }
  for (int i = tid; i < 3 * H; i += TPB) sW1[i / H][i % H] = W1g[i];
  for (int i = tid; i < H;  i += TPB) { sB1[i] = b1g[i]; sB2[i] = b2g[i]; }
  for (int i = tid; i < H2; i += TPB) sB3[i] = b3g[i];
  for (int i = tid; i < H2 * 2; i += TPB) sW4[i >> 1][i & 1] = W4g[i];
  if (tid < 2) sB4[tid] = b4g[tid];
  for (int i = tid; i < WAVES * SPW * 2; i += TPB) {
    ((float*)accb)[i]  = 0.0f;
    ((float*)dyacc)[i] = 0.0f;
  }
  __syncthreads();

  const float tval  = tg[0];
  const int   sbase = blockIdx.x * SPB + wave * SPW;

  // ================= stage y; gather per-lane sample regs =================
  ((float*)&xbuf[wave][0][0])[lane]      = yg[sbase * 2 + lane];
  ((float*)&xbuf[wave][0][0])[lane + 32] = yg[sbase * 2 + lane + 32];
  float ya[2][8], yb[2][8];
#pragma unroll
  for (int mt = 0; mt < 2; ++mt)
#pragma unroll
    for (int r = 0; r < 8; ++r) {
      ya[mt][r] = xbuf[wave][mt * 16 + hi * 8 + r][0];
      yb[mt][r] = xbuf[wave][mt * 16 + hi * 8 + r][1];
    }

  // ================= layer 1 (K=3, VALU) -> h1, silu' =================
  for (int t = 0; t < 8; ++t) {
    int n = t * 16 + lm;
    float w0 = sW1[0][n], w1 = sW1[1][n];
    float c0 = fmaf(sW1[2][n], tval, sB1[n]);
#pragma unroll
    for (int mt = 0; mt < 2; ++mt) {
      v8h sv;
#pragma unroll
      for (int r = 0; r < 8; ++r) {
        float z = fmaf(ya[mt][r], w0, fmaf(yb[mt][r], w1, c0));
        float h, sp;  silu_pair(z, h, sp);
        habuf[wave][mt * 16 + hi * 8 + r][n] = (_Float16)h;
        sv[r] = (_Float16)sp;
      }
      *(v8h*)&sp1b[wave][mt][t][lane][0] = sv;
    }
  }

  // ================= layer 2: z2 = h1 @ W2 (WMMA) =================
  v16h afr[2][4];
#pragma unroll
  for (int mt = 0; mt < 2; ++mt)
#pragma unroll
    for (int c = 0; c < 4; ++c)
      afr[mt][c] = load_frag(&habuf[wave][0][0], H, mt * 16 + lm, hi, c * 32);

  for (int t = 0; t < 8; ++t) {
    v8f ac[2] = {};
#pragma unroll
    for (int c = 0; c < 4; ++c) {
      v16h b = load_frag(&sW2T[0][0], H, t * 16 + lm, hi, c * 32);
      ac[0] = wmma16(afr[0][c], b, ac[0]);   // B reused for both m-tiles
      ac[1] = wmma16(afr[1][c], b, ac[1]);
    }
    int n = t * 16 + lm;
    float bn = sB2[n];
#pragma unroll
    for (int mt = 0; mt < 2; ++mt) {
      v8h sv;
#pragma unroll
      for (int r = 0; r < 8; ++r) {
        float h, sp;  silu_pair(ac[mt][r] + bn, h, sp);
        habuf[wave][mt * 16 + hi * 8 + r][n] = (_Float16)h;  // h2 overwrites h1
        sv[r] = (_Float16)sp;
      }
      *(v8h*)&sp2b[wave][mt][t][lane][0] = sv;
    }
  }

  // ===== layer 3: z3 = h2 @ W3 (WMMA); layer 4 fused into epilogue =====
#pragma unroll
  for (int mt = 0; mt < 2; ++mt)
#pragma unroll
    for (int c = 0; c < 4; ++c)
      afr[mt][c] = load_frag(&habuf[wave][0][0], H, mt * 16 + lm, hi, c * 32);

  float q0[2][8], q1[2][8];   // dy partials per row
#pragma unroll
  for (int mt = 0; mt < 2; ++mt)
#pragma unroll
    for (int r = 0; r < 8; ++r) { q0[mt][r] = 0.0f; q1[mt][r] = 0.0f; }

  for (int t = 0; t < 4; ++t) {
    v8f ac[2] = {};
#pragma unroll
    for (int c = 0; c < 4; ++c) {
      v16h b = load_frag(&sW3T[0][0], H, t * 16 + lm, hi, c * 32);
      ac[0] = wmma16(afr[0][c], b, ac[0]);
      ac[1] = wmma16(afr[1][c], b, ac[1]);
    }
    int n = t * 16 + lm;
    float bn = sB3[n];
    float w40 = sW4[n][0], w41 = sW4[n][1];
#pragma unroll
    for (int mt = 0; mt < 2; ++mt) {
      v8h sv;
#pragma unroll
      for (int r = 0; r < 8; ++r) {
        float h, sp;  silu_pair(ac[mt][r] + bn, h, sp);
        q0[mt][r] = fmaf(h, w40, q0[mt][r]);   // layer-4 partials
        q1[mt][r] = fmaf(h, w41, q1[mt][r]);
        sv[r] = (_Float16)sp;                  // h3 never materialized
      }
      *(v8h*)&sp3b[wave][mt][t][lane][0] = sv;
    }
  }
#pragma unroll
  for (int mt = 0; mt < 2; ++mt)
#pragma unroll
    for (int r = 0; r < 8; ++r) {
      atomicAdd(&dyacc[wave][mt * 16 + hi * 8 + r][0], q0[mt][r]);  // ds_add_f32
      atomicAdd(&dyacc[wave][mt * 16 + hi * 8 + r][1], q1[mt][r]);
    }

  // ================= stage v =================
  ((float*)&xbuf[wave][0][0])[lane]      = vg[sbase * 2 + lane];
  ((float*)&xbuf[wave][0][0])[lane + 32] = vg[sbase * 2 + lane + 32];
  float va[2][8], vb[2][8];
#pragma unroll
  for (int mt = 0; mt < 2; ++mt)
#pragma unroll
    for (int r = 0; r < 8; ++r) {
      va[mt][r] = xbuf[wave][mt * 16 + hi * 8 + r][0];
      vb[mt][r] = xbuf[wave][mt * 16 + hi * 8 + r][1];
    }

  // ===== backward layer 4: g3' = (v @ W4^T) * silu'(z3)  (VALU) =====
  for (int t = 0; t < 4; ++t) {
    int n = t * 16 + lm;
    float w40 = sW4[n][0], w41 = sW4[n][1];
#pragma unroll
    for (int mt = 0; mt < 2; ++mt) {
      v8h sp = *(const v8h*)&sp3b[wave][mt][t][lane][0];
#pragma unroll
      for (int r = 0; r < 8; ++r) {
        float g = fmaf(va[mt][r], w40, vb[mt][r] * w41) * (float)sp[r];
        habuf[wave][mt * 16 + hi * 8 + r][n] = (_Float16)g;
      }
    }
  }

  // ===== backward layer 3: g2 = g3' @ W3^T (WMMA), * silu'(z2) =====
  v16h gfr[2][4];
#pragma unroll
  for (int mt = 0; mt < 2; ++mt)
#pragma unroll
    for (int c = 0; c < 2; ++c)
      gfr[mt][c] = load_frag(&habuf[wave][0][0], H, mt * 16 + lm, hi, c * 32);

  for (int t = 0; t < 8; ++t) {
    v8f ac[2] = {};
#pragma unroll
    for (int c = 0; c < 2; ++c) {
      v16h b = load_frag(&sW3[0][0], H2, t * 16 + lm, hi, c * 32);
      ac[0] = wmma16(gfr[0][c], b, ac[0]);
      ac[1] = wmma16(gfr[1][c], b, ac[1]);
    }
    int n = t * 16 + lm;
#pragma unroll
    for (int mt = 0; mt < 2; ++mt) {
      v8h sp = *(const v8h*)&sp2b[wave][mt][t][lane][0];
#pragma unroll
      for (int r = 0; r < 8; ++r) {
        float g = ac[mt][r] * (float)sp[r];
        habuf[wave][mt * 16 + hi * 8 + r][n] = (_Float16)g;
      }
    }
  }

  // ===== backward layer 2: g1 = g2' @ W2^T (WMMA), * silu'(z1), reduce =====
#pragma unroll
  for (int mt = 0; mt < 2; ++mt)
#pragma unroll
    for (int c = 0; c < 4; ++c)
      gfr[mt][c] = load_frag(&habuf[wave][0][0], H, mt * 16 + lm, hi, c * 32);

  float s0[2][8], s1[2][8];
#pragma unroll
  for (int mt = 0; mt < 2; ++mt)
#pragma unroll
    for (int r = 0; r < 8; ++r) { s0[mt][r] = 0.0f; s1[mt][r] = 0.0f; }

  for (int t = 0; t < 8; ++t) {
    v8f ac[2] = {};
#pragma unroll
    for (int c = 0; c < 4; ++c) {
      v16h b = load_frag(&sW2[0][0], H, t * 16 + lm, hi, c * 32);
      ac[0] = wmma16(gfr[0][c], b, ac[0]);
      ac[1] = wmma16(gfr[1][c], b, ac[1]);
    }
    int n = t * 16 + lm;
    float w10 = sW1[0][n], w11 = sW1[1][n];
#pragma unroll
    for (int mt = 0; mt < 2; ++mt) {
      v8h sp = *(const v8h*)&sp1b[wave][mt][t][lane][0];
#pragma unroll
      for (int r = 0; r < 8; ++r) {
        float g = ac[mt][r] * (float)sp[r];
        s0[mt][r] = fmaf(g, w10, s0[mt][r]);
        s1[mt][r] = fmaf(g, w11, s1[mt][r]);
      }
    }
  }

#pragma unroll
  for (int mt = 0; mt < 2; ++mt)
#pragma unroll
    for (int r = 0; r < 8; ++r) {
      atomicAdd(&accb[wave][mt * 16 + hi * 8 + r][0], s0[mt][r]);  // ds_add_f32
      atomicAdd(&accb[wave][mt * 16 + hi * 8 + r][1], s1[mt][r]);
    }
  __syncthreads();

  // ================= outputs (all 32 lanes, no divergence) =================
#pragma unroll
  for (int mt = 0; mt < 2; ++mt)
    dyg[(sbase + mt * 16 + lm) * 2 + hi] = dyacc[wave][mt * 16 + lm][hi] + sB4[hi];
  {
    int m = lane;   // 32 samples per wave, one per lane
    float gv = accb[wave][m][0] * xbuf[wave][m][0]
             + accb[wave][m][1] * xbuf[wave][m][1];
    ndivg[sbase + m] = -gv;   // -div
  }
}

extern "C" void kernel_launch(void* const* d_in, const int* in_sizes, int n_in,
                              void* d_out, int out_size, void* d_ws, size_t ws_size,
                              hipStream_t stream) {
  (void)n_in; (void)out_size; (void)d_ws; (void)ws_size;
  const float* y  = (const float*)d_in[0];
  // d_in[1] = logp (unused by outputs)
  const float* t  = (const float*)d_in[2];
  const float* v  = (const float*)d_in[3];
  const float* W1 = (const float*)d_in[4];
  const float* b1 = (const float*)d_in[5];
  const float* W2 = (const float*)d_in[6];
  const float* b2 = (const float*)d_in[7];
  const float* W3 = (const float*)d_in[8];
  const float* b3 = (const float*)d_in[9];
  const float* W4 = (const float*)d_in[10];
  const float* b4 = (const float*)d_in[11];

  const int n = in_sizes[0] / 2;       // number of samples
  float* out  = (float*)d_out;
  float* dy   = out;                   // [n,2] flat
  float* ndiv = out + (size_t)n * 2;   // [n,1] flat

  const int blocks = n / SPB;
  hipLaunchKernelGGL(cnf_ode_wmma_kernel, dim3(blocks), dim3(TPB), 0, stream,
                     y, t, v, W1, b1, W2, b2, W3, b3, W4, b4, dy, ndiv);
}
